// polydentate_OMGNN_RNN_47425028882825
// MI455X (gfx1250) — compile-verified, matrix operand order checked by reading, not run on hardware
//
#include <hip/hip_runtime.h>
#include <hip/hip_bf16.h>

// D-MPNN (chemprop) on gfx1250.
// bf16 edge states + v_wmma_f32_16x16x32_bf16, f32 atomic segment-sum.
// ws layout (bytes):  H0 bf16 [0,153.6M) | Ha bf16 [153.6M,307.2M)
//                     Hb bf16 [307.2M,460.8M) | agg f32 [460.8M,512M)

#define NN   100000
#define EE   600000
#define HID  128

typedef __attribute__((ext_vector_type(16))) __bf16 v16bf;
typedef __attribute__((ext_vector_type(8)))  float  v8f;

union Frag { unsigned u[8]; v16bf v; };

__device__ __forceinline__ unsigned pack2(float lo, float hi) {
    union { __bf16 b[2]; unsigned u; } t;
    t.b[0] = (__bf16)lo; t.b[1] = (__bf16)hi;
    return t.u;
}
__device__ __forceinline__ unsigned short bf16bits(float v) {
    union { __bf16 b; unsigned short s; } t;
    t.b = (__bf16)v;
    return t.s;
}
__device__ __forceinline__ void unpack2(unsigned u, float& lo, float& hi) {
    union { unsigned u; __bf16 b[2]; } t;
    t.u = u;
    lo = (float)t.b[0]; hi = (float)t.b[1];
}

// fragment pair-index for slot j: k = (j<4 ? 2j : 2(j-4)+16) + 8*hi + 32*kc
__device__ __forceinline__ int frag_pj(int j, int hi16, int kc) {
    return (j & 3) + ((j >> 2) << 3) + (hi16 << 2) + kc * 16;
}

// ---------------- Kernel 1: H0 = relu([x[src];edge_attr] @ W_i + b_i) -------
#define KP1 80   // K padded 144 -> 160, as bf16 pairs

__global__ __launch_bounds__(256) void k_edge_init(
    const float* __restrict__ x, const float* __restrict__ ea,
    const float* __restrict__ Wi, const float* __restrict__ bi,
    const int* __restrict__ src,
    __bf16* __restrict__ H0, __bf16* __restrict__ H)
{
    __shared__ unsigned sW[128 * KP1];                 // W_i^T bf16 pairs, 40KB
    __shared__ unsigned sA[16 * KP1];                  // gathered rows, 5KB
    __shared__ float    sb[128];
    __shared__ __align__(16) unsigned short sD[16 * 128];  // D staging, 4KB
    const int tid = threadIdx.x;
    const int e0  = blockIdx.x * 16;

    // W_i^T staging: consecutive lanes read consecutive n -> coalesced
    for (int i = tid; i < 128 * KP1; i += 256) {
        int n = i & 127, p = i >> 7, k = 2 * p;
        float lo = (k     < 144) ? Wi[(long)k * 128 + n]       : 0.f;
        float hi = (k + 1 < 144) ? Wi[(long)(k + 1) * 128 + n] : 0.f;
        sW[n * KP1 + p] = pack2(lo, hi);
    }
    if (tid < 128) sb[tid] = bi[tid];
    const float2* x2  = (const float2*)x;
    const float2* ea2 = (const float2*)ea;
    for (int i = tid; i < 16 * KP1; i += 256) {
        int r = i / KP1, p = i % KP1, k = 2 * p;
        long e = e0 + r;
        long s = src[e];
        float lo, hi;
        if (k < 128)      { float2 v = x2[(s * 128 + k) >> 1];        lo = v.x; hi = v.y; }
        else if (k < 144) { float2 v = ea2[(e * 16 + (k - 128)) >> 1]; lo = v.x; hi = v.y; }
        else              { lo = 0.f; hi = 0.f; }
        sA[i] = pack2(lo, hi);
    }
    __syncthreads();

    const int lane = tid & 31, wave = tid >> 5;
    const int l16 = lane & 15, hi16 = lane >> 4;
    const int n0 = wave * 16;
    v8f c = {};
    for (int kc = 0; kc < 5; ++kc) {
        Frag a, b;
        #pragma unroll
        for (int j = 0; j < 8; ++j) {
            int pj = frag_pj(j, hi16, kc);
            a.u[j] = sA[l16 * KP1 + pj];
            b.u[j] = sW[(n0 + l16) * KP1 + pj];
        }
        c = __builtin_amdgcn_wmma_f32_16x16x32_bf16(false, a.v, false, b.v,
                                                    (short)0, c, false, false);
    }
    #pragma unroll
    for (int r = 0; r < 8; ++r) {
        int row = r + 8 * hi16, col = n0 + l16;
        float v = c[r] + sb[col];
        sD[row * 128 + col] = bf16bits(v > 0.f ? v : 0.f);
    }
    __syncthreads();
    // coalesced b128 row stores: 16 rows x 256B
    {
        int row = tid >> 4, ch = tid & 15;
        long e = e0 + row;
        uint4 vdat = ((const uint4*)sD)[row * 16 + ch];
        ((uint4*)(H0 + e * 128))[ch] = vdat;
        ((uint4*)(H  + e * 128))[ch] = vdat;
    }
}

// ---------------- Kernel 2: agg[dst[e]] += H[e]  (f32 atomics) --------------
__global__ __launch_bounds__(256) void k_scatter(
    const __bf16* __restrict__ H, const int* __restrict__ dst,
    float* __restrict__ agg)
{
    long idx = (long)blockIdx.x * 256 + threadIdx.x;   // over EE*64 pairs
    int e = (int)(idx >> 6);
    int p = (int)(idx & 63);
    const unsigned* Hu = (const unsigned*)H;
    float lo, hi;
    unpack2(Hu[idx], lo, hi);
    long d = dst[e];
    atomicAdd(&agg[d * 128 + 2 * p],     lo);
    atomicAdd(&agg[d * 128 + 2 * p + 1], hi);
}

// -------- Kernel 3: H' = relu(H0 + (agg[src]-H[rev]) @ W_h + b_h) -----------
__global__ __launch_bounds__(256) void k_update(
    const __bf16* __restrict__ H0, const float* __restrict__ agg,
    const __bf16* __restrict__ Hin, const float* __restrict__ Wh,
    const float* __restrict__ bh, const int* __restrict__ src,
    const int* __restrict__ rev, __bf16* __restrict__ Hout)
{
    __shared__ unsigned sW[128 * 64];   // W_h^T, 32KB
    __shared__ unsigned sA[16 * 64];    // message rows, 4KB
    __shared__ float    sb[128];
    __shared__ __align__(16) unsigned short sD[16 * 128];  // 4KB
    const int tid = threadIdx.x;
    const int e0  = blockIdx.x * 16;

    for (int i = tid; i < 128 * 64; i += 256) {
        int n = i & 127, p = i >> 7, k = 2 * p;   // coalesced on n
        sW[n * 64 + p] = pack2(Wh[(long)k * 128 + n], Wh[(long)(k + 1) * 128 + n]);
    }
    if (tid < 128) sb[tid] = bh[tid];
    const float2*   agg2 = (const float2*)agg;
    const unsigned* Hu   = (const unsigned*)Hin;
    for (int i = tid; i < 16 * 64; i += 256) {
        int r = i >> 6, p = i & 63, k = 2 * p;
        long e  = e0 + r;
        long s  = src[e];
        long re = rev[e];
        float2 av = agg2[(s * 128 + k) >> 1];
        float hl, hh;
        unpack2(Hu[(re * 128 + k) >> 1], hl, hh);
        sA[i] = pack2(av.x - hl, av.y - hh);
    }
    __syncthreads();

    const int lane = tid & 31, wave = tid >> 5;
    const int l16 = lane & 15, hi16 = lane >> 4;
    const int n0 = wave * 16;
    v8f c = {};
    for (int kc = 0; kc < 4; ++kc) {
        Frag a, b;
        #pragma unroll
        for (int j = 0; j < 8; ++j) {
            int pj = frag_pj(j, hi16, kc);
            a.u[j] = sA[l16 * 64 + pj];
            b.u[j] = sW[(n0 + l16) * 64 + pj];
        }
        c = __builtin_amdgcn_wmma_f32_16x16x32_bf16(false, a.v, false, b.v,
                                                    (short)0, c, false, false);
    }
    #pragma unroll
    for (int r = 0; r < 8; ++r) {
        int row = r + 8 * hi16, col = n0 + l16;
        long e = e0 + row;
        float v = c[r] + sb[col] + (float)H0[e * 128 + col];
        sD[row * 128 + col] = bf16bits(v > 0.f ? v : 0.f);
    }
    __syncthreads();
    {
        int row = tid >> 4, ch = tid & 15;
        long e = e0 + row;
        ((uint4*)(Hout + e * 128))[ch] = ((const uint4*)sD)[row * 16 + ch];
    }
}

// -------- Kernel 4: out = relu([x ; where(sum(Mn)==0,x,Mn)] @ W_o + b_o) ----
__global__ __launch_bounds__(256) void k_final(
    const float* __restrict__ x, const float* __restrict__ agg,
    const float* __restrict__ Wo, const float* __restrict__ bo,
    float* __restrict__ out)
{
    __shared__ unsigned sW[128 * 64];   // half of W_o^T at a time, 32KB
    __shared__ unsigned sA[16 * 128];   // [x ; Mn] rows (K=256), 8KB
    __shared__ float    sb[128];
    __shared__ float    rs[16];
    __shared__ __align__(16) float sO[16 * 128];   // 8KB output staging
    const int tid = threadIdx.x;
    const int i0  = blockIdx.x * 16;

    if (tid < 128) sb[tid] = bo[tid];
    if (tid < 16)  rs[tid] = 0.f;
    __syncthreads();
    // row sums of Mn (untouched rows are exactly 0 from memset -> == 0 test OK)
    for (int i = tid; i < 16 * 128; i += 256) {
        int r = i >> 7, k = i & 127;
        atomicAdd(&rs[r], agg[(long)(i0 + r) * 128 + k]);
    }
    __syncthreads();
    const float2* x2 = (const float2*)x;
    const float2* a2 = (const float2*)agg;
    for (int i = tid; i < 16 * 128; i += 256) {
        int r = i >> 7, p = i & 127, k = 2 * p;
        long node = i0 + r;
        float2 v;
        if (k < 128) v = x2[(node * 128 + k) >> 1];
        else {
            int kk = k - 128;
            const float2* sp = (rs[r] == 0.f) ? x2 : a2;
            v = sp[(node * 128 + kk) >> 1];
        }
        sA[i] = pack2(v.x, v.y);
    }

    const int lane = tid & 31, wave = tid >> 5;
    const int l16 = lane & 15, hi16 = lane >> 4;
    const int n0 = wave * 16;
    v8f c = {};
    for (int half = 0; half < 2; ++half) {
        __syncthreads();   // also makes sA/rs visible before first half
        for (int i = tid; i < 128 * 64; i += 256) {
            int n = i & 127, p = i >> 7;          // coalesced on n
            int k = 2 * (p + half * 64);
            sW[n * 64 + p] = pack2(Wo[(long)k * 128 + n], Wo[(long)(k + 1) * 128 + n]);
        }
        __syncthreads();
        for (int kc = 0; kc < 4; ++kc) {
            Frag a, b;
            #pragma unroll
            for (int j = 0; j < 8; ++j) {
                int pj = frag_pj(j, hi16, kc);
                a.u[j] = sA[l16 * 128 + half * 64 + pj];
                b.u[j] = sW[(n0 + l16) * 64 + pj];
            }
            c = __builtin_amdgcn_wmma_f32_16x16x32_bf16(false, a.v, false, b.v,
                                                        (short)0, c, false, false);
        }
    }
    #pragma unroll
    for (int r = 0; r < 8; ++r) {
        int row = r + 8 * hi16, col = n0 + l16;
        float v = c[r] + sb[col];
        sO[row * 128 + col] = v > 0.f ? v : 0.f;
    }
    __syncthreads();
    // coalesced f32 row stores: 16 rows x 512B, 2x b128 per thread
    for (int i = tid; i < 16 * 32; i += 256) {
        int row = i >> 5, ch = i & 31;
        long node = i0 + row;
        ((float4*)(out + node * 128))[ch] = ((const float4*)sO)[row * 32 + ch];
    }
}

// ---------------------------------------------------------------------------
extern "C" void kernel_launch(void* const* d_in, const int* in_sizes, int n_in,
                              void* d_out, int out_size, void* d_ws, size_t ws_size,
                              hipStream_t stream) {
    const float* x    = (const float*)d_in[0];
    const float* ea   = (const float*)d_in[1];
    const float* Wi   = (const float*)d_in[2];
    const float* bi   = (const float*)d_in[3];
    const float* Wh   = (const float*)d_in[4];
    const float* bh   = (const float*)d_in[5];
    const float* Wo   = (const float*)d_in[6];
    const float* bo   = (const float*)d_in[7];
    const int*   eidx = (const int*)d_in[8];
    const int*   rev  = (const int*)d_in[9];
    const int*   src  = eidx;
    const int*   dst  = eidx + EE;

    char* ws = (char*)d_ws;
    const size_t Hbytes = (size_t)EE * HID * sizeof(__bf16);     // 153.6 MB
    __bf16* H0  = (__bf16*)(ws);
    __bf16* Ha  = (__bf16*)(ws + Hbytes);
    __bf16* Hb  = (__bf16*)(ws + 2 * Hbytes);
    float*  agg = (float*)(ws + 3 * Hbytes);
    const size_t aggBytes = (size_t)NN * HID * sizeof(float);    // 51.2 MB
    float* out = (float*)d_out;

    const int gE  = EE / 16;            // 37500
    const int gS  = (EE * 64) / 256;    // 150000
    const int gN  = NN / 16;            // 6250

    k_edge_init<<<gE, 256, 0, stream>>>(x, ea, Wi, bi, src, H0, Ha);

    // depth iteration 1
    hipMemsetAsync(agg, 0, aggBytes, stream);
    k_scatter<<<gS, 256, 0, stream>>>(Ha, dst, agg);
    k_update <<<gE, 256, 0, stream>>>(H0, agg, Ha, Wh, bh, src, rev, Hb);

    // depth iteration 2
    hipMemsetAsync(agg, 0, aggBytes, stream);
    k_scatter<<<gS, 256, 0, stream>>>(Hb, dst, agg);
    k_update <<<gE, 256, 0, stream>>>(H0, agg, Hb, Wh, bh, src, rev, Ha);

    // final node aggregation + readout
    hipMemsetAsync(agg, 0, aggBytes, stream);
    k_scatter<<<gS, 256, 0, stream>>>(Ha, dst, agg);
    k_final  <<<gN, 256, 0, stream>>>(x, agg, Wo, bo, out);
}
